// HPTEncoder_3874060501345
// MI455X (gfx1250) — matrix operand order repaired and checked by reading, not compiled, not run on hardware
//
#include <hip/hip_runtime.h>

// Longformer sliding-window + global attention for MI455X (gfx1250).
// All matmuls via v_wmma_f32_16x16x32_bf16 (fp32 accumulate); softmax in fp32.

typedef __attribute__((ext_vector_type(16))) __bf16 v16bf;
typedef __attribute__((ext_vector_type(8)))  __bf16 v8bf;
typedef __attribute__((ext_vector_type(8)))  float  v8f;

#define S_LEN 4096
#define HID   768
#define NHEAD 12
#define HDIM  64
#define WIN   256
#define GLB   64
#define CHK   256     // chunk size (== W)
#define BANDW 768     // 3*CHK band columns
#define NLOG  832     // BANDW + GLB logit columns per query
#define NEGV  (-1.0e9f)

// ---------------- WMMA fragment helpers (contiguous k order, used for both A and B) --------------
// A (16x32 bf16): lane<16 holds row=lane, K {k0..k0+15};
//                 lane>=16 holds row=lane-16, K {k0+16..k0+31}.
__device__ __forceinline__ v16bf cat8(v8bf lo, v8bf hi) {
  v16bf r;
#pragma unroll
  for (int i = 0; i < 8; ++i) { r[i] = lo[i]; r[i + 8] = hi[i]; }
  return r;
}

__device__ __forceinline__ v16bf zero16() {
  v16bf r;
#pragma unroll
  for (int i = 0; i < 16; ++i) r[i] = (__bf16)0.0f;
  return r;
}

__device__ __forceinline__ v16bf loadA32(const __bf16* rowPtr, int k0, int lane) {
  const int off = (lane & 16) ? 16 : 0;
  v8bf lo = *reinterpret_cast<const v8bf*>(rowPtr + k0 + off);
  v8bf hi = *reinterpret_cast<const v8bf*>(rowPtr + k0 + 8 + off);
  return cat8(lo, hi);
}

// B (32x16 bf16): lane holds col=lane%16, K run of 16 starting at k0 (+16 for
// upper lane half). Caller passes pointer to that contiguous 16-element run.
__device__ __forceinline__ v16bf loadB16(const __bf16* p16) {
  const v8bf* p = reinterpret_cast<const v8bf*>(p16);
  return cat8(p[0], p[1]);
}

__device__ __forceinline__ v8f wmma_bf16(v16bf a, v16bf b, v8f c) {
  return __builtin_amdgcn_wmma_f32_16x16x32_bf16(false, a, false, b,
                                                 (short)0, c, false, false);
}

// ---------------- conversion kernels ----------------------------------------
__global__ void cvt_bf16_kernel(const float* __restrict__ src,
                                __bf16* __restrict__ dst, int n) {
  int i = blockIdx.x * blockDim.x + threadIdx.x;
  if (i < n) dst[i] = (__bf16)src[i];
}

// dst[n][k] = src[k][n]  (HID x HID), bf16 transposed weights
__global__ void cvtT_bf16_kernel(const float* __restrict__ src,
                                 __bf16* __restrict__ dst) {
  int nIdx = blockIdx.x * 16 + threadIdx.x;
  int k    = blockIdx.y * 16 + threadIdx.y;
  dst[(size_t)nIdx * HID + k] = (__bf16)src[(size_t)k * HID + nIdx];
}

// ---------------- projection GEMM: out = (X @ W + b) * scale ----------------
// A = Xbf [M x 768], BT = W^T bf16 [768 x 768]. Each wave owns a 64x16 output
// tile (4 M-subtiles sharing one B fragment per K chunk -> 2.5 loads/WMMA).
// transposed==0: dst[h][m][d] (ldM = rows/head); transposed==1: dst[h][d][m].
__global__ __launch_bounds__(256) void proj_gemm_kernel(
    const __bf16* __restrict__ A, const __bf16* __restrict__ BT,
    const float* __restrict__ bias, float scale, __bf16* __restrict__ dst,
    int ldM, int transposed) {
  const int wave = threadIdx.x >> 5, lane = threadIdx.x & 31;
  const int l16 = lane & 15, hiH = (lane & 16) ? 1 : 0;
  const int mGroup = blockIdx.x;               // 64 rows per group
  const int nTile = blockIdx.y * 8 + wave;
  const int nIdx = nTile * 16 + l16;
  const __bf16* ar0 = A + (size_t)(mGroup * 64 + l16) * HID;
  const __bf16* br  = BT + (size_t)nIdx * HID;
  v8f acc0 = {}, acc1 = {}, acc2 = {}, acc3 = {};
#pragma unroll 2
  for (int kc = 0; kc < HID / 32; ++kc) {
    int k0 = kc * 32;
    v16bf b = loadB16(br + k0 + (hiH ? 16 : 0));
    acc0 = wmma_bf16(loadA32(ar0 + (size_t)0  * HID, k0, lane), b, acc0);
    acc1 = wmma_bf16(loadA32(ar0 + (size_t)16 * HID, k0, lane), b, acc1);
    acc2 = wmma_bf16(loadA32(ar0 + (size_t)32 * HID, k0, lane), b, acc2);
    acc3 = wmma_bf16(loadA32(ar0 + (size_t)48 * HID, k0, lane), b, acc3);
  }
  const float bb = bias[nIdx];
  const int h = nIdx >> 6, d = nIdx & 63;
  v8f accs[4] = {acc0, acc1, acc2, acc3};
#pragma unroll
  for (int s = 0; s < 4; ++s) {
    const int rBase = mGroup * 64 + s * 16 + (hiH ? 8 : 0);
    if (!transposed) {
#pragma unroll
      for (int i = 0; i < 8; ++i) {
        float v = (accs[s][i] + bb) * scale;
        dst[((size_t)h * ldM + rBase + i) * HDIM + d] = (__bf16)v;
      }
    } else {
      v8bf pk;                                  // rows contiguous in [h][d][s]
#pragma unroll
      for (int i = 0; i < 8; ++i) pk[i] = (__bf16)((accs[s][i] + bb) * scale);
      *reinterpret_cast<v8bf*>(dst + ((size_t)h * HDIM + d) * ldM + rBase) = pk;
    }
  }
}

// ---------------- band + global-column attention -----------------------------
// Block = 256 thr (8 waves) per (strip, chunk, head): 16 query rows, 832 cols.
__global__ __launch_bounds__(256) void band_attn_kernel(
    const __bf16* __restrict__ qb, const __bf16* __restrict__ kb,
    const __bf16* __restrict__ vTb, float* __restrict__ out) {
  __shared__ float lds_s[16 * NLOG];          // 53 KB: fp32 scores, then bf16 P
  const int strip = blockIdx.x;               // 0..15 (16-row strip in chunk)
  const int n     = blockIdx.y;               // chunk 0..15
  const int h     = blockIdx.z;               // head
  const int tid = threadIdx.x;
  const int wave = tid >> 5, lane = tid & 31;
  const int l16 = lane & 15, hiH = (lane & 16) ? 1 : 0;

  // Phase 1: logits into LDS. Q fragments held in registers for the strip.
  const int qrow = n * CHK + strip * 16 + l16;
  const __bf16* qr = qb + ((size_t)h * S_LEN + qrow) * HDIM;
  const v16bf a0 = loadA32(qr, 0, lane);
  const v16bf a1 = loadA32(qr, 32, lane);

  for (int t = wave; t < 52; t += 8) {        // 48 band tiles + 4 global tiles
    v8f acc = {};
    if (t < 48) {
      const int j0 = t * 16;
      const int keyBase = n * CHK - CHK + j0; // multiple of 16: whole tile in/out
      v16bf b0, b1;
      if (keyBase >= 0 && keyBase < S_LEN) {
        const __bf16* kr = kb + ((size_t)h * S_LEN + keyBase + l16) * HDIM;
        b0 = loadB16(kr + (hiH ? 16 : 0));
        b1 = loadB16(kr + 32 + (hiH ? 16 : 0));
      } else { b0 = zero16(); b1 = zero16(); }
      acc = wmma_bf16(a0, b0, acc);
      acc = wmma_bf16(a1, b1, acc);
      const int jcol = j0 + l16;
      const int keyAbs = keyBase + l16;
      const bool colOk = (keyBase >= 0) && (keyBase < S_LEN) && (keyAbs >= GLB);
#pragma unroll
      for (int i = 0; i < 8; ++i) {
        int r = i + (hiH ? 8 : 0);
        int iq = strip * 16 + r;
        int rel = jcol - CHK - iq;
        bool ok = colOk && (rel <= WIN) && (rel >= -WIN);
        lds_s[r * NLOG + jcol] = ok ? acc[i] : NEGV;
      }
    } else {
      const int g0 = (t - 48) * 16;           // global key columns 0..63
      const __bf16* kr = kb + ((size_t)h * S_LEN + g0 + l16) * HDIM;
      v16bf b0 = loadB16(kr + (hiH ? 16 : 0));
      v16bf b1 = loadB16(kr + 32 + (hiH ? 16 : 0));
      acc = wmma_bf16(a0, b0, acc);
      acc = wmma_bf16(a1, b1, acc);
#pragma unroll
      for (int i = 0; i < 8; ++i) {
        int r = i + (hiH ? 8 : 0);
        lds_s[r * NLOG + BANDW + g0 + l16] = acc[i];
      }
    }
  }
  __syncthreads();

  // Phase 2: joint softmax over 832 columns; 16 lanes per row, shuffle reduce.
  const int row = tid >> 4;
  const int sub = tid & 15;
  float vals[52];
  float mx = -3.0e38f;
#pragma unroll
  for (int i = 0; i < 52; ++i) {
    vals[i] = lds_s[row * NLOG + sub + i * 16];
    mx = fmaxf(mx, vals[i]);
  }
#pragma unroll
  for (int off = 8; off >= 1; off >>= 1) mx = fmaxf(mx, __shfl_xor(mx, off, 16));
  float sum = 0.f;
#pragma unroll
  for (int i = 0; i < 52; ++i) { vals[i] = __expf(vals[i] - mx); sum += vals[i]; }
#pragma unroll
  for (int off = 8; off >= 1; off >>= 1) sum += __shfl_xor(sum, off, 16);
  const float inv = 1.0f / sum;
  __syncthreads();                             // all fp32 reads done
  __bf16* pl = (__bf16*)lds_s;                 // reuse LDS as bf16 P [16][832]
#pragma unroll
  for (int i = 0; i < 52; ++i)
    pl[row * NLOG + sub + i * 16] = (__bf16)(vals[i] * inv);
  __syncthreads();

  // Phase 3: out = P[16x832] @ Vcat[832x64]; waves 0..3 own d-tiles (uniform).
  if (wave < 4) {
    const int d = wave * 16 + l16;
    const __bf16* vrow = vTb + ((size_t)h * HDIM + d) * S_LEN; // v^T [d][s]
    const __bf16* pl2 = (const __bf16*)lds_s;
    v8f acc = {};
    for (int kc = 0; kc < 26; ++kc) {          // 24 band + 2 global K-chunks
      int k0 = kc * 32;
      v16bf a = loadA32(pl2 + l16 * NLOG, k0, lane);
      v16bf b;
      if (k0 < BANDW) {
        int keyBase = n * CHK - CHK + k0 + (hiH ? 16 : 0);
        b = (keyBase >= 0 && keyBase < S_LEN) ? loadB16(vrow + keyBase)
                                              : zero16();   // masked P == 0
      } else {
        b = loadB16(vrow + (k0 - BANDW) + (hiH ? 16 : 0));  // global V cols
      }
      acc = wmma_bf16(a, b, acc);
    }
    const int sBase = n * CHK + strip * 16;
#pragma unroll
    for (int i = 0; i < 8; ++i) {
      int r = i + (hiH ? 8 : 0);
      out[(size_t)(sBase + r) * HID + h * HDIM + d] = acc[i];
    }
  }
}

// ---------------- global-query rows ------------------------------------------
__global__ __launch_bounds__(256) void gq_scores_kernel(
    const __bf16* __restrict__ qg, const __bf16* __restrict__ kg,
    float* __restrict__ gs) {
  const int wave = threadIdx.x >> 5, lane = threadIdx.x & 31;
  const int l16 = lane & 15, hiH = (lane & 16) ? 1 : 0;
  const int h = blockIdx.x;
  const int rTile = wave & 3;
  const int cTile = blockIdx.y * 2 + (wave >> 2);
  const __bf16* ar = qg + ((size_t)h * GLB + rTile * 16 + l16) * HDIM;
  const __bf16* br = kg + ((size_t)h * S_LEN + cTile * 16 + l16) * HDIM;
  v8f acc = {};
  acc = wmma_bf16(loadA32(ar, 0, lane),  loadB16(br + (hiH ? 16 : 0)), acc);
  acc = wmma_bf16(loadA32(ar, 32, lane), loadB16(br + 32 + (hiH ? 16 : 0)), acc);
  const int col = cTile * 16 + l16;
#pragma unroll
  for (int i = 0; i < 8; ++i) {
    int r = rTile * 16 + i + (hiH ? 8 : 0);
    gs[((size_t)h * GLB + r) * S_LEN + col] = acc[i];
  }
}

__global__ __launch_bounds__(256) void gq_softmax_kernel(
    const float* __restrict__ gs, __bf16* __restrict__ gp) {
  __shared__ float red[256];
  const int r = blockIdx.x;                    // 0..NHEAD*GLB-1
  const int tid = threadIdx.x;
  float vals[16];
  float mx = -3.0e38f;
#pragma unroll
  for (int i = 0; i < 16; ++i) {
    vals[i] = gs[(size_t)r * S_LEN + tid + i * 256];
    mx = fmaxf(mx, vals[i]);
  }
  red[tid] = mx; __syncthreads();
  for (int s = 128; s > 0; s >>= 1) {
    if (tid < s) red[tid] = fmaxf(red[tid], red[tid + s]);
    __syncthreads();
  }
  mx = red[0]; __syncthreads();
  float sum = 0.f;
#pragma unroll
  for (int i = 0; i < 16; ++i) { vals[i] = __expf(vals[i] - mx); sum += vals[i]; }
  red[tid] = sum; __syncthreads();
  for (int s = 128; s > 0; s >>= 1) {
    if (tid < s) red[tid] += red[tid + s];
    __syncthreads();
  }
  const float inv = 1.0f / red[0];
#pragma unroll
  for (int i = 0; i < 16; ++i)
    gp[(size_t)r * S_LEN + tid + i * 256] = (__bf16)(vals[i] * inv);
}

// out rows < GLB: wave owns 2 row-tiles sharing the B fragment (K=4096 GEMM).
__global__ __launch_bounds__(256) void gq_out_kernel(
    const __bf16* __restrict__ gp, const __bf16* __restrict__ vgT,
    float* __restrict__ out) {
  const int wave = threadIdx.x >> 5, lane = threadIdx.x & 31;
  const int l16 = lane & 15, hiH = (lane & 16) ? 1 : 0;
  const int h = blockIdx.x;
  const int dTile = wave & 3;
  const int rBase = (wave >> 2) * 2;           // rTiles rBase, rBase+1
  const __bf16* ar0 = gp + ((size_t)h * GLB + rBase * 16 + l16) * S_LEN;
  const __bf16* ar1 = ar0 + (size_t)16 * S_LEN;
  const int d = dTile * 16 + l16;
  const __bf16* br = vgT + ((size_t)h * HDIM + d) * S_LEN;
  v8f acc0 = {}, acc1 = {};
#pragma unroll 2
  for (int kc = 0; kc < S_LEN / 32; ++kc) {
    int k0 = kc * 32;
    v16bf b = loadB16(br + k0 + (hiH ? 16 : 0));
    acc0 = wmma_bf16(loadA32(ar0, k0, lane), b, acc0);
    acc1 = wmma_bf16(loadA32(ar1, k0, lane), b, acc1);
  }
#pragma unroll
  for (int i = 0; i < 8; ++i) {
    int r0 = (rBase + 0) * 16 + i + (hiH ? 8 : 0);
    int r1 = (rBase + 1) * 16 + i + (hiH ? 8 : 0);
    out[(size_t)r0 * HID + h * HDIM + d] = acc0[i];
    out[(size_t)r1 * HID + h * HDIM + d] = acc1[i];
  }
}

// ---------------- launch -----------------------------------------------------
extern "C" void kernel_launch(void* const* d_in, const int* in_sizes, int n_in,
                              void* d_out, int out_size, void* d_ws, size_t ws_size,
                              hipStream_t stream) {
  (void)in_sizes; (void)n_in; (void)out_size; (void)ws_size;
  const float* X   = (const float*)d_in[0];
  const float* Wq  = (const float*)d_in[1];  const float* bq  = (const float*)d_in[2];
  const float* Wk  = (const float*)d_in[3];  const float* bk  = (const float*)d_in[4];
  const float* Wv  = (const float*)d_in[5];  const float* bv  = (const float*)d_in[6];
  const float* Wqg = (const float*)d_in[7];  const float* bqg = (const float*)d_in[8];
  const float* Wkg = (const float*)d_in[9];  const float* bkg = (const float*)d_in[10];
  const float* Wvg = (const float*)d_in[11]; const float* bvg = (const float*)d_in[12];
  float* out = (float*)d_out;

  char* ws = (char*)d_ws;
  size_t off = 0;
  auto alloc = [&](size_t bytes) -> void* {
    void* p = ws + off;
    off = (off + bytes + 255) & ~(size_t)255;
    return p;
  };
  const size_t xBytes = (size_t)S_LEN * HID * 2;
  const size_t wBytes = (size_t)HID * HID * 2;
  const size_t hBytes = (size_t)NHEAD * S_LEN * HDIM * 2;

  __bf16* Xbf  = (__bf16*)alloc(xBytes);
  __bf16* WTq  = (__bf16*)alloc(wBytes);
  __bf16* WTk  = (__bf16*)alloc(wBytes);
  __bf16* WTv  = (__bf16*)alloc(wBytes);
  __bf16* WTqg = (__bf16*)alloc(wBytes);
  __bf16* WTkg = (__bf16*)alloc(wBytes);
  __bf16* WTvg = (__bf16*)alloc(wBytes);
  __bf16* qbf  = (__bf16*)alloc(hBytes);                       // [h][s][d], scaled
  __bf16* kbf  = (__bf16*)alloc(hBytes);                       // [h][s][d]
  __bf16* vT   = (__bf16*)alloc(hBytes);                       // [h][d][s]
  __bf16* qgbf = (__bf16*)alloc((size_t)NHEAD * GLB * HDIM * 2);
  __bf16* kgbf = (__bf16*)alloc(hBytes);                       // [h][s][d]
  __bf16* vgT  = (__bf16*)alloc(hBytes);                       // [h][d][s]
  float*  gs   = (float*)alloc((size_t)NHEAD * GLB * S_LEN * 4);
  __bf16* gp   = (__bf16*)alloc((size_t)NHEAD * GLB * S_LEN * 2);

  const float scale = 0.125f;                  // 1/sqrt(64)

  // 1) precision conversion / weight transpose
  const int nX = S_LEN * HID;
  cvt_bf16_kernel<<<nX / 256, 256, 0, stream>>>(X, Xbf, nX);
  dim3 tb(16, 16), tg(HID / 16, HID / 16);
  cvtT_bf16_kernel<<<tg, tb, 0, stream>>>(Wq,  WTq);
  cvtT_bf16_kernel<<<tg, tb, 0, stream>>>(Wk,  WTk);
  cvtT_bf16_kernel<<<tg, tb, 0, stream>>>(Wv,  WTv);
  cvtT_bf16_kernel<<<tg, tb, 0, stream>>>(Wqg, WTqg);
  cvtT_bf16_kernel<<<tg, tb, 0, stream>>>(Wkg, WTkg);
  cvtT_bf16_kernel<<<tg, tb, 0, stream>>>(Wvg, WTvg);

  // 2) QKV projections (WMMA), 64x16 tile per wave
  dim3 pg(S_LEN / 64, HID / 16 / 8);
  proj_gemm_kernel<<<pg, 256, 0, stream>>>(Xbf, WTq,  bq,  scale, qbf,  S_LEN, 0);
  proj_gemm_kernel<<<pg, 256, 0, stream>>>(Xbf, WTk,  bk,  1.0f,  kbf,  S_LEN, 0);
  proj_gemm_kernel<<<pg, 256, 0, stream>>>(Xbf, WTv,  bv,  1.0f,  vT,   S_LEN, 1);
  proj_gemm_kernel<<<pg, 256, 0, stream>>>(Xbf, WTkg, bkg, 1.0f,  kgbf, S_LEN, 0);
  proj_gemm_kernel<<<pg, 256, 0, stream>>>(Xbf, WTvg, bvg, 1.0f,  vgT,  S_LEN, 1);
  dim3 pgg(GLB / 64, HID / 16 / 8);
  proj_gemm_kernel<<<pgg, 256, 0, stream>>>(Xbf, WTqg, bqg, scale, qgbf, GLB, 0);

  // 3) sliding-window + global-column attention (writes all of out)
  band_attn_kernel<<<dim3(CHK / 16, S_LEN / CHK, NHEAD), 256, 0, stream>>>(
      qbf, kbf, vT, out);

  // 4) global-query rows (overwrite rows < GLB)
  gq_scores_kernel<<<dim3(NHEAD, S_LEN / 32), 256, 0, stream>>>(qgbf, kgbf, gs);
  gq_softmax_kernel<<<NHEAD * GLB, 256, 0, stream>>>(gs, gp);
  gq_out_kernel<<<NHEAD, 256, 0, stream>>>(gp, vgT, out);
}
